// Model_10797547782832
// MI455X (gfx1250) — matrix operand order repaired and checked by reading, not compiled, not run on hardware
//
#include <hip/hip_runtime.h>
#include <hip/hip_bf16.h>

#define Bsz 2048
#define Tsz 512
#define Hsz 256          // hidden; 4H = 1024 gate columns
#define BT 16            // batch rows per block
#define NWAVE 16
#define THREADS (NWAVE * 32)
#define HPAD 264         // padded bf16 elements per ldsH row (16B rotation per row)

typedef __bf16 bf16_t;
typedef __attribute__((ext_vector_type(16))) __bf16 v16bf;
typedef __attribute__((ext_vector_type(8)))  float  v8f;
typedef __attribute__((ext_vector_type(4)))  unsigned int uint4v;

// Global-address-space pointer types: keep weight-stream loads on the
// GLOBAL_LOAD path (LOADcnt only) instead of FLAT (LOADcnt + DScnt).
typedef const __attribute__((address_space(1))) bf16_t* gbf_ptr;
typedef const __attribute__((address_space(1))) uint4v* g4_ptr;

struct Frag { uint4v u[2]; };   // 32 bytes / lane = one 16x16x32 bf16 fragment

__device__ __forceinline__ v16bf frag_cast(const Frag& f) {
    union { Frag f; v16bf v; } u; u.f = f; return u.v;
}

// ---------------------------------------------------------------------------
// One-time weight prep: W_hh (f32, [4H][H] row-major; gates = h @ W_hh.T so
// B[k][n] = W_hh[n][k]) -> bf16, packed per-lane in the exact WMMA B-fragment
// order for v_wmma_f32_16x16x32_bf16.
// Fragment (nt, ks): lane l holds column n = nt*16 + (l&15);
//   k-base = ks*32 + (l<16 ? 0 : 8); halves = {kb..kb+7, kb+16..kb+23}.
// Packed at wsB[((nt*8 + ks)*32 + lane)*16 .. +15]  (32B/lane, coalesced).
// ---------------------------------------------------------------------------
__global__ void prep_whh(const float* __restrict__ W_hh, bf16_t* __restrict__ wsB) {
    int idx  = blockIdx.x * blockDim.x + threadIdx.x;   // 64*8*32 = 16384
    int nt   = idx >> 8;
    int ks   = (idx >> 5) & 7;
    int lane = idx & 31;
    int col  = nt * 16 + (lane & 15);
    int kb   = ks * 32 + ((lane & 16) ? 8 : 0);
    const float* src = W_hh + (size_t)col * Hsz;
    bf16_t* dst = wsB + ((size_t)((nt * 8 + ks) * 32 + lane)) * 16;
#pragma unroll
    for (int i = 0; i < 8; ++i) dst[i]     = (bf16_t)src[kb + i];
#pragma unroll
    for (int i = 0; i < 8; ++i) dst[8 + i] = (bf16_t)src[kb + 16 + i];
}

__device__ __forceinline__ float sigmoidf_(float v) {
    return 1.0f / (1.0f + __expf(-v));
}

// ---------------------------------------------------------------------------
// LSTM scan. Block = 16 batch rows, 16 waves. Wave w owns gate columns
// [16w,16w+16) of each gate (N-tiles w, 16+w, 32+w, 48+w), so the i/f/g/o
// accumulator tiles are lane-aligned and the c/h update is lane-local.
// Weights stream from L2 every step (512 KB/WG/step, L2-resident) with an
// explicit 2-deep software pipeline; h is re-shared per step via LDS (bf16).
// ---------------------------------------------------------------------------
__global__ void __launch_bounds__(THREADS, 1)
lstm_scan(const float* __restrict__ x,     const float* __restrict__ W_ih,
          const float* __restrict__ b_ih,  const float* __restrict__ b_hh,
          const float* __restrict__ W_lin, const float* __restrict__ b_lin,
          const bf16_t* __restrict__ wsB,  float* __restrict__ out) {
    __shared__ __align__(16) float  ldsX[BT * Tsz];    // 32 KB x-tile
    __shared__ __align__(16) bf16_t ldsH[BT * HPAD];   // 8.25 KB h (bf16, padded)
    __shared__             float  ldsOut[BT];

    const int tid  = threadIdx.x;
    const int lane = tid & 31;
    const int w    = tid >> 5;        // wave 0..15
    const int hi   = lane >> 4;       // half-wave: M rows 0..7 / 8..15
    const int ln   = lane & 15;       // column within tile
    const int b0   = blockIdx.x * BT;

    // Stage this block's x slice [BT][T] into LDS (coalesced over t).
    for (int i = tid; i < BT * Tsz; i += THREADS)
        ldsX[i] = x[(size_t)(b0 + (i >> 9)) * Tsz + (i & 511)];
    for (int i = tid; i < BT * HPAD; i += THREADS)
        ldsH[i] = (bf16_t)0.0f;                        // h0 = 0
    if (tid < BT) ldsOut[tid] = 0.0f;

    // Per-lane constants: gate column n = j*H + 16w + ln for j = i,f,g,o.
    float wih[4], bias[4];
#pragma unroll
    for (int j = 0; j < 4; ++j) {
        int n   = j * Hsz + w * 16 + ln;
        wih[j]  = W_ih[n];
        bias[j] = b_ih[n] + b_hh[n];
    }
    const float wl   = W_lin[w * 16 + ln];
    const float blin = b_lin[0];

    // Per-lane weight base: fragment (nt=j*16+w, ks) lives at
    //   wsB + (j*16+w)*4096 + ks*512 + lane*16   (halves)
    // => wp + j*131072 + ks*1024                 (bytes), immediates only.
    const gbf_ptr wp0 = (gbf_ptr)(wsB + ((size_t)w * 4096 + (size_t)lane * 16));

    // Per-lane A base in ldsH (halves; + ks*32 per k-step).
    const int arow = ln * HPAD + hi * 8;

    v8f c = {};                            // cell-state tile in WMMA C layout
    __syncthreads();

    for (int t = 0; t < Tsz; ++t) {
        // Opaque global pointer: blocks LICM from hoisting weight loads out of
        // the t-loop while keeping the AS(1) GLOBAL_LOAD path (LOADcnt only).
        gbf_ptr wp = wp0;
        asm volatile("" : "+v"(wp));

        // acc = bias + x_t * W_ih  (outer product, lane-local in C layout)
        float xv[8];
#pragma unroll
        for (int v = 0; v < 8; ++v) xv[v] = ldsX[(hi * 8 + v) * Tsz + t];
        v8f acc[4];
#pragma unroll
        for (int j = 0; j < 4; ++j)
#pragma unroll
            for (int v = 0; v < 8; ++v)
                acc[j][v] = fmaf(xv[v], wih[j], bias[j]);

        // ---- gates += h @ W_hh.T : K=256, 8 k-steps, 2-deep pipeline ----
        Frag bfr[2][4];
        Frag afr[2];
        {   // prologue: k-step 0
            const uint4v* ap = (const uint4v*)&ldsH[arow];
            afr[0].u[0] = ap[0];
            afr[0].u[1] = ap[2];            // +16 halves = ap[2]
#pragma unroll
            for (int j = 0; j < 4; ++j) {
                g4_ptr bp = (g4_ptr)(wp + (size_t)j * 65536);
                bfr[0][j].u[0] = bp[0];
                bfr[0][j].u[1] = bp[1];
            }
        }
#pragma unroll
        for (int ks = 0; ks < 8; ++ks) {
            const int cur = ks & 1, nxt = cur ^ 1;
            if (ks < 7) {   // prefetch k-step ks+1 while computing ks
                const uint4v* ap = (const uint4v*)&ldsH[arow + (ks + 1) * 32];
                afr[nxt].u[0] = ap[0];
                afr[nxt].u[1] = ap[2];
#pragma unroll
                for (int j = 0; j < 4; ++j) {
                    g4_ptr bp = (g4_ptr)(wp + (size_t)j * 65536 + (ks + 1) * 512);
                    bfr[nxt][j].u[0] = bp[0];
                    bfr[nxt][j].u[1] = bp[1];
                }
            }
            v16bf A = frag_cast(afr[cur]);
#pragma unroll
            for (int j = 0; j < 4; ++j)
                acc[j] = __builtin_amdgcn_wmma_f32_16x16x32_bf16(
                    false, A, false, frag_cast(bfr[cur][j]),
                    (short)0, acc[j], false, false);
            __builtin_amdgcn_sched_barrier(0);   // fence each k-step stage
        }
        __syncthreads();   // all waves done reading ldsH; ldsOut reset visible

        // Elementwise LSTM cell update (lane-local; m = hi*8+v, n = 16w+ln).
        float hval[8];
#pragma unroll
        for (int v = 0; v < 8; ++v) {
            float ig = sigmoidf_(acc[0][v]);
            float fg = sigmoidf_(acc[1][v]);
            float gg = tanhf(acc[2][v]);
            float og = sigmoidf_(acc[3][v]);
            float cn = fg * c[v] + ig * gg;
            c[v]     = cn;
            hval[v]  = og * tanhf(cn);
        }

        // Publish h (bf16) for next step + per-row output partial h·W_lin.
        const int col = w * 16 + ln;
        float part[8];
#pragma unroll
        for (int v = 0; v < 8; ++v) {
            ldsH[(hi * 8 + v) * HPAD + col] = (bf16_t)hval[v];
            part[v] = hval[v] * wl;
        }
#pragma unroll
        for (int v = 0; v < 8; ++v) {
#pragma unroll
            for (int off = 1; off < 16; off <<= 1)
                part[v] += __shfl_xor(part[v], off, 32);
        }
        if (ln == 0) {
#pragma unroll
            for (int v = 0; v < 8; ++v)
                atomicAdd(&ldsOut[hi * 8 + v], part[v]);   // ds_add_f32
        }
        __syncthreads();

        if (w == 0 && lane < BT) {
            out[(size_t)(b0 + lane) * Tsz + t] = ldsOut[lane] + blin;
            ldsOut[lane] = 0.0f;       // next step's adds gated by next barrier
        }
    }
}

extern "C" void kernel_launch(void* const* d_in, const int* in_sizes, int n_in,
                              void* d_out, int out_size, void* d_ws, size_t ws_size,
                              hipStream_t stream) {
    const float* x     = (const float*)d_in[0];
    const float* W_ih  = (const float*)d_in[1];
    const float* W_hh  = (const float*)d_in[2];
    const float* b_ih  = (const float*)d_in[3];
    const float* b_hh  = (const float*)d_in[4];
    const float* W_lin = (const float*)d_in[5];
    const float* b_lin = (const float*)d_in[6];
    bf16_t* wsB = (bf16_t*)d_ws;            // 64*8*32*16 bf16 = 512 KB
    float*  out = (float*)d_out;

    prep_whh<<<64, 256, 0, stream>>>(W_hh, wsB);
    lstm_scan<<<Bsz / BT, THREADS, 0, stream>>>(x, W_ih, b_ih, b_hh,
                                                W_lin, b_lin, wsB, out);
}
